// PDConvLSTM_10436770529628
// MI455X (gfx1250) — matrix operand order repaired and checked by reading, not compiled, not run on hardware
//
#include <hip/hip_runtime.h>
#include <stdint.h>

// ---------------------------------------------------------------------------
// PDConvLSTM forward for MI455X (gfx1250), wave32 + WMMA bf16.
//
// Roofline: ~787 GFLOP in ConvLSTM gate convs vs ~100 MB of traffic ->
// compute bound -> v_wmma_f32_16x16x32_bf16 with f32 accumulation.
// Weights are packed once to bf16 in a tap-major K order so every 32-wide
// WMMA k-step sits inside one 3x3 tap; B fragments then read 16 contiguous
// channels from an LDS tile (channel-fastest layout) as two b128 DS loads.
// ---------------------------------------------------------------------------

typedef __bf16 bf16;
typedef __attribute__((ext_vector_type(8)))  bf16  bf16x8;
typedef __attribute__((ext_vector_type(16))) bf16  bf16x16;
typedef __attribute__((ext_vector_type(8)))  float f32x8;

#define NB   64     // batch
#define NT   24     // timesteps
#define IND  25     // input channels
#define HID  64     // hidden channels
#define PIX  625    // 25*25

static __device__ __forceinline__ float sigf(float x) {
  return 1.0f / (1.0f + __expf(-x));
}

static __device__ __forceinline__ bf16x16 cat16(bf16x8 lo, bf16x8 hi) {
  return __builtin_shufflevector(lo, hi, 0,1,2,3,4,5,6,7,8,9,10,11,12,13,14,15);
}

// ---------------------------------------------------------------------------
// LayerNorm over the 25-channel axis + convert to bf16.
// x: (B,T,25,25,25) f32 -> xn: same shape bf16. One thread per (b,t,pixel).
// ---------------------------------------------------------------------------
__global__ __launch_bounds__(256)
void ln_kernel(const float* __restrict__ x, const float* __restrict__ g,
               const float* __restrict__ bb, bf16* __restrict__ xn) {
  int tid = blockIdx.x * 256 + threadIdx.x;
  if (tid >= NB * NT * PIX) return;
  int bt  = tid / PIX;
  int pix = tid % PIX;
  size_t base = (size_t)bt * IND * PIX + pix;
  float s = 0.f, sq = 0.f;
  for (int c = 0; c < IND; ++c) {
    float v = x[base + (size_t)c * PIX];
    s += v; sq += v * v;
  }
  float mu  = s * (1.0f / IND);
  float var = sq * (1.0f / IND) - mu * mu;
  float inv = rsqrtf(var + 1e-5f);
  for (int c = 0; c < IND; ++c) {
    float v = x[base + (size_t)c * PIX];
    xn[base + (size_t)c * PIX] = (bf16)((v - mu) * inv * g[c] + bb[c]);
  }
}

// ---------------------------------------------------------------------------
// Weight packing: w (M, Cin, 3, 3) f32 -> bf16 [M][9*Cpad], k = tap*Cpad + ic.
// ---------------------------------------------------------------------------
__global__ __launch_bounds__(256)
void pack_w_kernel(const float* __restrict__ w, bf16* __restrict__ wp,
                   int M, int Cin, int Cpad) {
  int total = M * 9 * Cpad;
  int i = blockIdx.x * 256 + threadIdx.x;
  if (i >= total) return;
  int Kp  = 9 * Cpad;
  int m   = i / Kp;
  int k   = i % Kp;
  int tap = k / Cpad;
  int ic  = k % Cpad;
  int ky  = tap / 3, kx = tap % 3;
  float v = 0.f;
  if (ic < Cin) v = w[(((size_t)m * Cin + ic) * 3 + ky) * 3 + kx];
  wp[i] = (bf16)v;
}

__global__ __launch_bounds__(256)
void f2bf_kernel(const float* __restrict__ a, bf16* __restrict__ o, int n) {
  int i = blockIdx.x * 256 + threadIdx.x;
  if (i < n) o[i] = (bf16)a[i];
}

// ---------------------------------------------------------------------------
// ConvLSTM gate step: implicit-GEMM (M=256 gates, K=9*96, N=one output row
// padded 25->32) via v_wmma_f32_16x16x32_bf16, then fused LSTM cell update.
// Workgroup = one (batch, row). 512 threads = 16 waves; wave w owns M-tile w
// and both N-tiles. LDS: halo tile (96ch x 3rows x 40cols, channel-fastest)
// + 256x32 f32 z-buffer for the cross-wave gate combine.
// ---------------------------------------------------------------------------
__global__ __launch_bounds__(512)
void gate_kernel(const bf16* __restrict__ xn, int t,
                 const bf16* __restrict__ hprev,
                 const bf16* __restrict__ wpack,   // [256][9*96]
                 const float* __restrict__ bias,   // [256]
                 float* __restrict__ cbuf,         // [B][64][625]
                 bf16* __restrict__ hnext,         // [B][64][625]
                 int dil) {
  __shared__ bf16  s_in[3 * 40 * 96];   // [(row3*40+cc)*96 + ch]
  __shared__ float s_z[256 * 32];

  const int b = blockIdx.x / 25;
  const int y = blockIdx.x % 25;
  const int tid = threadIdx.x;

  // Stage input + hidden halo into LDS (channels 89..95 zero-padded).
  for (int i = tid; i < 3 * 40 * 96; i += 512) {
    int ch = i % 96;
    int rc = i / 96;
    int r3 = rc / 40;
    int cc = rc % 40;
    int xx = cc - dil;
    int yy = y + dil * (r3 - 1);
    float v = 0.f;
    if (xx >= 0 && xx < 25 && yy >= 0 && yy < 25) {
      if (ch < IND)
        v = (float)xn[(((size_t)b * NT + t) * IND + ch) * PIX + yy * 25 + xx];
      else if (ch < IND + HID)
        v = (float)hprev[((size_t)b * HID + (ch - IND)) * PIX + yy * 25 + xx];
    }
    s_in[i] = (bf16)v;
  }
  __syncthreads();

  const int w    = tid >> 5;     // wave id == M tile (0..15)
  const int lane = tid & 31;
  const int half = lane >> 4;
  const int ln16 = lane & 15;

  const bf16* wr = wpack + (size_t)(16 * w + ln16) * (9 * 96);
  __builtin_prefetch(wr, 0, 3);

  f32x8 acc0 = {}; f32x8 acc1 = {};
  for (int s = 0; s < 27; ++s) {       // K = 864 = 27 * 32
    const int kbase = 32 * s;
    const int tap   = s / 3;           // each k-step inside one tap
    const int chb   = 32 * (s % 3);
    const int ky    = tap / 3, kx = tap % 3;
    // A fragment (contiguous 16 K per half-wave, matching B)
    bf16x8 alo = *(const bf16x8*)(wr + kbase + half * 16);
    bf16x8 ahi = *(const bf16x8*)(wr + kbase + half * 16 + 8);
    bf16x16 a  = cat16(alo, ahi);
    // B fragments: 16 contiguous channels from LDS (two b128 DS loads)
    {
      int xin = ln16 + dil * kx;                    // N-tile 0
      const bf16* bp = &s_in[(ky * 40 + xin) * 96 + chb + 16 * half];
      bf16x16 bfrag = cat16(*(const bf16x8*)bp, *(const bf16x8*)(bp + 8));
      acc0 = __builtin_amdgcn_wmma_f32_16x16x32_bf16(
          false, a, false, bfrag, (short)0, acc0, false, false);
    }
    {
      int xin = 16 + ln16 + dil * kx;               // N-tile 1
      const bf16* bp = &s_in[(ky * 40 + xin) * 96 + chb + 16 * half];
      bf16x16 bfrag = cat16(*(const bf16x8*)bp, *(const bf16x8*)(bp + 8));
      acc1 = __builtin_amdgcn_wmma_f32_16x16x32_bf16(
          false, a, false, bfrag, (short)0, acc1, false, false);
    }
  }

  // Spill accumulators to z-buffer (D layout: lane half selects M+8).
  for (int r = 0; r < 8; ++r) {
    int m = 16 * w + r + 8 * half;
    s_z[m * 32 + ln16]      = acc0[r];
    s_z[m * 32 + 16 + ln16] = acc1[r];
  }
  __syncthreads();

  // Fused LSTM cell: gates i,f,o,g are channel blocks [0:64)..[192:256).
  for (int i = tid; i < HID * 25; i += 512) {
    int hc = i / 25;
    int xx = i % 25;
    float zi = s_z[(hc      ) * 32 + xx] + bias[hc];
    float zf = s_z[( 64 + hc) * 32 + xx] + bias[ 64 + hc];
    float zo = s_z[(128 + hc) * 32 + xx] + bias[128 + hc];
    float zg = s_z[(192 + hc) * 32 + xx] + bias[192 + hc];
    size_t idx = ((size_t)b * HID + hc) * PIX + y * 25 + xx;
    float c = cbuf[idx];
    c = sigf(zf) * c + sigf(zi) * tanhf(zg);
    cbuf[idx]  = c;
    hnext[idx] = (bf16)(sigf(zo) * tanhf(c));
  }
}

// ---------------------------------------------------------------------------
// Head conv: concat(h1,h2) (128 ch) -> 64 ch, 3x3, relu.  Same implicit GEMM
// (M=64, K=9*128=1152, N=row of 25 padded to 32). 256 threads = 8 waves:
// wave w -> (mtile = w&3, ntile = w>>2). Epilogue writes f32 directly.
// ---------------------------------------------------------------------------
__global__ __launch_bounds__(256)
void headconv_kernel(const bf16* __restrict__ h1, const bf16* __restrict__ h2,
                     const bf16* __restrict__ wpack,   // [64][9*128]
                     const float* __restrict__ bias,   // [64]
                     float* __restrict__ out) {        // [B][64][625]
  __shared__ bf16 s_in[3 * 40 * 128];

  const int b = blockIdx.x / 25;
  const int y = blockIdx.x % 25;
  const int tid = threadIdx.x;

  for (int i = tid; i < 3 * 40 * 128; i += 256) {
    int ch = i % 128;
    int rc = i / 128;
    int r3 = rc / 40;
    int cc = rc % 40;
    int xx = cc - 1;
    int yy = y + (r3 - 1);
    float v = 0.f;
    if (xx >= 0 && xx < 25 && yy >= 0 && yy < 25) {
      const bf16* src = (ch < HID) ? h1 : h2;
      int c2 = (ch < HID) ? ch : ch - HID;
      v = (float)src[((size_t)b * HID + c2) * PIX + yy * 25 + xx];
    }
    s_in[i] = (bf16)v;
  }
  __syncthreads();

  const int w    = tid >> 5;
  const int lane = tid & 31;
  const int half = lane >> 4;
  const int ln16 = lane & 15;
  const int mtile = w & 3;
  const int ntile = w >> 2;

  const bf16* wr = wpack + (size_t)(16 * mtile + ln16) * (9 * 128);

  f32x8 acc = {};
  for (int s = 0; s < 36; ++s) {       // K = 1152 = 36 * 32
    const int kbase = 32 * s;
    const int tap   = s / 4;
    const int chb   = 32 * (s % 4);
    const int ky    = tap / 3, kx = tap % 3;
    bf16x8 alo = *(const bf16x8*)(wr + kbase + half * 16);
    bf16x8 ahi = *(const bf16x8*)(wr + kbase + half * 16 + 8);
    bf16x16 a  = cat16(alo, ahi);
    int xin = 16 * ntile + ln16 + kx;
    const bf16* bp = &s_in[(ky * 40 + xin) * 128 + chb + 16 * half];
    bf16x16 bfrag = cat16(*(const bf16x8*)bp, *(const bf16x8*)(bp + 8));
    acc = __builtin_amdgcn_wmma_f32_16x16x32_bf16(
        false, a, false, bfrag, (short)0, acc, false, false);
  }

  for (int r = 0; r < 8; ++r) {
    int oc = 16 * mtile + r + 8 * half;
    int xo = 16 * ntile + ln16;
    if (xo < 25)
      out[((size_t)b * HID + oc) * PIX + y * 25 + xo] =
          fmaxf(acc[r] + bias[oc], 0.f);
  }
}

// ---------------------------------------------------------------------------
// 2x2 maxpool (25->12 VALID) + channel-major flatten to bf16 fc1 input.
// ---------------------------------------------------------------------------
__global__ __launch_bounds__(256)
void pool_kernel(const float* __restrict__ conv, bf16* __restrict__ pooled) {
  int tid = blockIdx.x * 256 + threadIdx.x;
  if (tid >= NB * HID * 144) return;
  int b   = tid / (HID * 144);
  int rem = tid % (HID * 144);
  int c   = rem / 144;
  int pp  = rem % 144;
  int py  = pp / 12, px = pp % 12;
  size_t base = ((size_t)b * HID + c) * PIX + (2 * py) * 25 + 2 * px;
  float m = fmaxf(fmaxf(conv[base], conv[base + 1]),
                  fmaxf(conv[base + 25], conv[base + 26]));
  pooled[(size_t)b * 9216 + c * 144 + pp] = (bf16)m;
}

// ---------------------------------------------------------------------------
// fc1: 64x9216 @ 9216x64 -> 64x64, WMMA GEMM in a single workgroup.
// 512 threads = 16 waves = 4 M-tiles x 4 N-tiles. Fully contiguous 16B loads.
// ---------------------------------------------------------------------------
__global__ __launch_bounds__(512)
void fc1_kernel(const bf16* __restrict__ wf1,     // [64][9216] bf16
                const bf16* __restrict__ pooled,  // [64][9216] bf16
                const float* __restrict__ bias,   // [64]
                float* __restrict__ out) {        // [64 batch][64]
  const int tid  = threadIdx.x;
  const int w    = tid >> 5;
  const int lane = tid & 31;
  const int half = lane >> 4;
  const int ln16 = lane & 15;
  const int mtile = w & 3;
  const int ntile = w >> 2;

  const bf16* wr = wf1    + (size_t)(16 * mtile + ln16) * 9216;
  const bf16* br = pooled + (size_t)(16 * ntile + ln16) * 9216;

  f32x8 acc = {};
  for (int s = 0; s < 288; ++s) {      // K = 9216 = 288 * 32
    const int kbase = 32 * s;
    bf16x8 alo = *(const bf16x8*)(wr + kbase + half * 16);
    bf16x8 ahi = *(const bf16x8*)(wr + kbase + half * 16 + 8);
    bf16x16 a  = cat16(alo, ahi);
    const bf16* bp = br + kbase + 16 * half;   // B: contiguous 16 per lane
    bf16x16 bfrag  = cat16(*(const bf16x8*)bp, *(const bf16x8*)(bp + 8));
    acc = __builtin_amdgcn_wmma_f32_16x16x32_bf16(
        false, a, false, bfrag, (short)0, acc, false, false);
  }
  for (int r = 0; r < 8; ++r) {
    int m = 16 * mtile + r + 8 * half;   // output feature
    int n = 16 * ntile + ln16;           // batch
    out[n * 64 + m] = fmaxf(acc[r] + bias[m], 0.f);
  }
}

// ---------------------------------------------------------------------------
// fc2 -> relu -> fc3 -> log_softmax. Tiny: 64 threads, one per batch row.
// ---------------------------------------------------------------------------
__global__ __launch_bounds__(64)
void tail_kernel(const float* __restrict__ fc1o,
                 const float* __restrict__ w2, const float* __restrict__ b2,
                 const float* __restrict__ w3, const float* __restrict__ b3,
                 float* __restrict__ out) {
  int b = threadIdx.x;
  if (b >= NB) return;
  float a2[32];
  for (int o = 0; o < 32; ++o) {
    float s = b2[o];
    for (int k = 0; k < 64; ++k) s += w2[o * 64 + k] * fc1o[b * 64 + k];
    a2[o] = fmaxf(s, 0.f);
  }
  float a3[2];
  for (int o = 0; o < 2; ++o) {
    float s = b3[o];
    for (int k = 0; k < 32; ++k) s += w3[o * 32 + k] * a2[k];
    a3[o] = s;
  }
  float mx  = fmaxf(a3[0], a3[1]);
  float lse = mx + __logf(__expf(a3[0] - mx) + __expf(a3[1] - mx));
  out[b * 2 + 0] = a3[0] - lse;
  out[b * 2 + 1] = a3[1] - lse;
}

// ---------------------------------------------------------------------------
extern "C" void kernel_launch(void* const* d_in, const int* in_sizes, int n_in,
                              void* d_out, int out_size, void* d_ws, size_t ws_size,
                              hipStream_t stream) {
  (void)in_sizes; (void)n_in; (void)out_size; (void)ws_size;
  const float* x    = (const float*)d_in[0];
  const float* ln_g = (const float*)d_in[1];
  const float* ln_b = (const float*)d_in[2];
  const float* w1   = (const float*)d_in[3];
  const float* b1   = (const float*)d_in[4];
  const float* w2   = (const float*)d_in[5];
  const float* b2   = (const float*)d_in[6];
  const float* wc   = (const float*)d_in[7];
  const float* bc   = (const float*)d_in[8];
  const float* wf1  = (const float*)d_in[9];
  const float* bf1  = (const float*)d_in[10];
  const float* wf2  = (const float*)d_in[11];
  const float* bf2  = (const float*)d_in[12];
  const float* wf3  = (const float*)d_in[13];
  const float* bf3  = (const float*)d_in[14];
  float* out = (float*)d_out;

  // Carve workspace (~103 MB).
  uint8_t* p = (uint8_t*)d_ws;
  auto carve = [&](size_t bytes) -> void* {
    void* r = (void*)p;
    p += (bytes + 255) & ~(size_t)255;
    return r;
  };
  const size_t N_XN  = (size_t)NB * NT * IND * PIX;   // 24,000,000
  const size_t N_H   = (size_t)NB * HID * PIX;        //  2,560,000
  bf16*  xn   = (bf16*)carve(N_XN * 2);
  bf16*  w1p  = (bf16*)carve((size_t)256 * 864 * 2);
  bf16*  w2p  = (bf16*)carve((size_t)256 * 864 * 2);
  bf16*  wcp  = (bf16*)carve((size_t)64 * 1152 * 2);
  bf16*  wf1p = (bf16*)carve((size_t)64 * 9216 * 2);
  bf16*  h1a  = (bf16*)carve(N_H * 2);
  bf16*  h1b  = (bf16*)carve(N_H * 2);
  bf16*  h2a  = (bf16*)carve(N_H * 2);
  bf16*  h2b  = (bf16*)carve(N_H * 2);
  float* c1   = (float*)carve(N_H * 4);
  float* c2   = (float*)carve(N_H * 4);
  float* cvo  = (float*)carve(N_H * 4);
  bf16*  pool = (bf16*)carve((size_t)NB * 9216 * 2);
  float* f1o  = (float*)carve((size_t)NB * 64 * 4);

  // Zero-init recurrent state every call (deterministic, capture-safe).
  hipMemsetAsync(h1a, 0, N_H * 2, stream);
  hipMemsetAsync(h2a, 0, N_H * 2, stream);
  hipMemsetAsync(c1,  0, N_H * 4, stream);
  hipMemsetAsync(c2,  0, N_H * 4, stream);

  // Preprocess: LayerNorm->bf16, weight packing.
  ln_kernel<<<(NB * NT * PIX + 255) / 256, 256, 0, stream>>>(x, ln_g, ln_b, xn);
  pack_w_kernel<<<(256 * 864 + 255) / 256, 256, 0, stream>>>(w1, w1p, 256, 89, 96);
  pack_w_kernel<<<(256 * 864 + 255) / 256, 256, 0, stream>>>(w2, w2p, 256, 89, 96);
  pack_w_kernel<<<(64 * 1152 + 255) / 256, 256, 0, stream>>>(wc, wcp, 64, 128, 128);
  f2bf_kernel<<<(64 * 9216 + 255) / 256, 256, 0, stream>>>(wf1, wf1p, 64 * 9216);

  // 24 timesteps x 2 ConvLSTM branches (dil 1 and 2), h double-buffered.
  bf16 *h1c = h1a, *h1n = h1b, *h2c = h2a, *h2n = h2b;
  for (int t = 0; t < NT; ++t) {
    gate_kernel<<<NB * 25, 512, 0, stream>>>(xn, t, h1c, w1p, b1, c1, h1n, 1);
    gate_kernel<<<NB * 25, 512, 0, stream>>>(xn, t, h2c, w2p, b2, c2, h2n, 2);
    bf16* tmp;
    tmp = h1c; h1c = h1n; h1n = tmp;
    tmp = h2c; h2c = h2n; h2n = tmp;
  }

  // Head: conv+relu -> maxpool/flatten -> fc1 (WMMA) -> fc2/fc3/log_softmax.
  headconv_kernel<<<NB * 25, 256, 0, stream>>>(h1c, h2c, wcp, bc, cvo);
  pool_kernel<<<(NB * HID * 144 + 255) / 256, 256, 0, stream>>>(cvo, pool);
  fc1_kernel<<<1, 512, 0, stream>>>(wf1p, pool, bf1, f1o);
  tail_kernel<<<1, 64, 0, stream>>>(f1o, wf2, bf2, wf3, bf3, out);
}
